// SplitSoftmaxWithLoss_12695923327404
// MI455X (gfx1250) — matrix operand order repaired and checked by reading, not compiled, not run on hardware
//
#include <hip/hip_runtime.h>
#include <hip/hip_bf16.h>
#include <stdint.h>

typedef __attribute__((ext_vector_type(16))) _Float16 v16h;
typedef __attribute__((ext_vector_type(8)))  _Float16 v8h;
typedef __attribute__((ext_vector_type(8)))  float    v8f;

#define IN_F         1024
#define NCLASS       50257
#define CUT0         4000
#define CUT1         20000
#define NROWS        8192
#define ROWS_PER_BLK 64
#define AROW         516    // dwords per LDS row: 512 data + pad; 16B-aligned rows, 4-bank lane spread

#define CAT16(a,b) __builtin_shufflevector(a, b, 0,1,2,3,4,5,6,7,8,9,10,11,12,13,14,15)

__device__ __forceinline__ uint32_t pk2(_Float16 a, _Float16 b) {
  union { _Float16 f; uint16_t u; } x, y;
  x.f = a; y.f = b;
  return (uint32_t)x.u | ((uint32_t)y.u << 16);
}

// ---------------- K0: split fp32 weights into f16 hi/lo planes -------------
__global__ __launch_bounds__(256) void split_weight_kernel(
    const float* __restrict__ w, uint32_t* __restrict__ whi32,
    uint32_t* __restrict__ wlo32, int n4)
{
  int i = blockIdx.x * blockDim.x + threadIdx.x;
  if (i >= n4) return;
  float4 x = ((const float4*)w)[i];
  float xs[4] = {x.x, x.y, x.z, x.w};
  _Float16 h[4], l[4];
#pragma unroll
  for (int j = 0; j < 4; ++j) {
    h[j] = (_Float16)xs[j];
    l[j] = (_Float16)(xs[j] - (float)h[j]);
  }
  uint2 uh, ul;
  uh.x = pk2(h[0], h[1]); uh.y = pk2(h[2], h[3]);
  ul.x = pk2(l[0], l[1]); ul.y = pk2(l[2], l[3]);
  ((uint2*)whi32)[i] = uh;
  ((uint2*)wlo32)[i] = ul;
}

// ---------------- K1: per-row target logit + 2 tail-vector logits ----------
__global__ __launch_bounds__(256) void row_dots_kernel(
    const float* __restrict__ input, const int* __restrict__ target,
    const float* __restrict__ weight, const float* __restrict__ bias,
    const float* __restrict__ tailv, const float* __restrict__ tailb,
    float* __restrict__ tl, float* __restrict__ tv0, float* __restrict__ tv1)
{
  __shared__ float red[3][8];
  const int r = blockIdx.x;
  const int t = target[r];
  const float4* xr = (const float4*)(input + (size_t)r * IN_F);
  const float4* wr = (const float4*)(weight + (size_t)t * IN_F);
  const float4* t0 = (const float4*)(tailv);
  const float4* t1 = (const float4*)(tailv + IN_F);

  float4 x4 = xr[threadIdx.x];
  float4 w4 = wr[threadIdx.x];
  float4 a4 = t0[threadIdx.x];
  float4 b4 = t1[threadIdx.x];
  float s0 = x4.x*w4.x + x4.y*w4.y + x4.z*w4.z + x4.w*w4.w;
  float s1 = x4.x*a4.x + x4.y*a4.y + x4.z*a4.z + x4.w*a4.w;
  float s2 = x4.x*b4.x + x4.y*b4.y + x4.z*b4.z + x4.w*b4.w;
#pragma unroll
  for (int d = 1; d < 32; d <<= 1) {
    s0 += __shfl_xor(s0, d, 32);
    s1 += __shfl_xor(s1, d, 32);
    s2 += __shfl_xor(s2, d, 32);
  }
  const int wave = threadIdx.x >> 5, lane = threadIdx.x & 31;
  if (lane == 0) { red[0][wave] = s0; red[1][wave] = s1; red[2][wave] = s2; }
  __syncthreads();
  if (threadIdx.x == 0) {
    float a = 0.f, b = 0.f, c = 0.f;
#pragma unroll
    for (int i = 0; i < 8; ++i) { a += red[0][i]; b += red[1][i]; c += red[2][i]; }
    tl[r]  = a + bias[t];
    tv0[r] = b + tailb[0];
    tv1[r] = c + tailb[1];
  }
}

// ---------------- K2: fused WMMA GEMM + segmented online logsumexp ---------
// Fast path: weights pre-split to f16 hi/lo, A pre-shuffled to fragment order
// in LDS -> inner loop is pure ds_load/global_load + v_wmma. Three independent
// accumulator chains (hi*hi, lo*hi, hi*lo) keep the WMMA pipe busy.
__global__ __launch_bounds__(256) void lse_gemm_split_kernel(
    const float* __restrict__ input,
    const _Float16* __restrict__ whi, const _Float16* __restrict__ wlo,
    const float* __restrict__ bias,
    float* __restrict__ segM, float* __restrict__ segS)
{
  extern __shared__ uint32_t smem[];
  uint32_t* AHI = smem;                          // [64][AROW]
  uint32_t* ALO = smem + ROWS_PER_BLK * AROW;    // [64][AROW]
  float* mbuf = (float*)(ALO + ROWS_PER_BLK * AROW); // [2 slots][64 rows][M,S]

  const int rowbase  = blockIdx.x * ROWS_PER_BLK;
  const int seg      = blockIdx.y;
  const int segStart = (seg == 0) ? 0    : (seg == 1 ? CUT0 : CUT1);
  const int segEnd   = (seg == 0) ? CUT0 : (seg == 1 ? CUT1 : NCLASS);
  const int nTiles   = (segEnd - segStart + 15) >> 4;

  // Stage A (64 rows x 1024) as f16 hi/lo planes in WMMA-A fragment dword order:
  // per 32-k block, dwords in natural k order (lane half h reads k 16h..16h+15)
  for (int i = threadIdx.x; i < ROWS_PER_BLK * (IN_F / 2); i += blockDim.x) {
    int r  = i >> 9;           // row
    int d  = i & 511;          // dword within row
    int blk = d >> 4, dl = d & 15;
    int jl = dl;
    const float* src = input + (size_t)(rowbase + r) * IN_F + blk * 32 + dl * 2;
    float x0 = src[0], x1 = src[1];
    _Float16 h0 = (_Float16)x0, h1 = (_Float16)x1;
    _Float16 l0 = (_Float16)(x0 - (float)h0), l1 = (_Float16)(x1 - (float)h1);
    AHI[r * AROW + blk * 16 + jl] = pk2(h0, h1);
    ALO[r * AROW + blk * 16 + jl] = pk2(l0, l1);
  }
  __syncthreads();

  const int wave  = threadIdx.x >> 5;
  const int lane  = threadIdx.x & 31;
  const int rs    = wave >> 1;    // row subtile 0..3 (16 rows each)
  const int cs    = wave & 1;     // class slot 0..1
  const int half_ = lane >> 4;
  const int ln    = lane & 15;

  const uint32_t arow = (uint32_t)(rs * 16 + ln) * AROW + half_ * 8;

  float runM[8], runS[8];
#pragma unroll
  for (int v = 0; v < 8; ++v) { runM[v] = -1e30f; runS[v] = 0.f; }

  for (int t = cs; t < nTiles; t += 2) {
    const int cbase = segStart + (t << 4);
    const int c  = cbase + ln;
    const int cc = (c < NCLASS) ? c : (NCLASS - 1);
    const _Float16* bh = whi + (size_t)cc * IN_F + 16 * half_;
    const _Float16* bl = wlo + (size_t)cc * IN_F + 16 * half_;

    v8f accH = {0.f, 0.f, 0.f, 0.f, 0.f, 0.f, 0.f, 0.f};
    v8f accX = {0.f, 0.f, 0.f, 0.f, 0.f, 0.f, 0.f, 0.f};
    v8f accY = {0.f, 0.f, 0.f, 0.f, 0.f, 0.f, 0.f, 0.f};

#pragma unroll 4
    for (int kb = 0; kb < IN_F; kb += 32) {
      const v8h* ah = (const v8h*)(AHI + arow + (kb >> 1));
      const v8h* al = (const v8h*)(ALO + arow + (kb >> 1));
      v16h ahi = CAT16(ah[0], ah[1]);
      v16h alo = CAT16(al[0], al[1]);
      const v8h* bhp = (const v8h*)(bh + kb);
      const v8h* blp = (const v8h*)(bl + kb);
      v16h bhi = CAT16(bhp[0], bhp[1]);
      v16h blo = CAT16(blp[0], blp[1]);
      accX = __builtin_amdgcn_wmma_f32_16x16x32_f16(false, alo, false, bhi, (short)0, accX, false, false);
      accY = __builtin_amdgcn_wmma_f32_16x16x32_f16(false, ahi, false, blo, (short)0, accY, false, false);
      accH = __builtin_amdgcn_wmma_f32_16x16x32_f16(false, ahi, false, bhi, (short)0, accH, false, false);
    }

    const bool  valid = (c < segEnd);
    const float b = valid ? bias[c] : 0.f;
#pragma unroll
    for (int v = 0; v < 8; ++v) {
      float x = valid ? (accH[v] + (accX[v] + accY[v]) + b) : -INFINITY;
      float m = x;
#pragma unroll
      for (int d = 1; d < 16; d <<= 1) m = fmaxf(m, __shfl_xor(m, d, 32));
      float e = __expf(x - m);
      float s = e;
#pragma unroll
      for (int d = 1; d < 16; d <<= 1) s += __shfl_xor(s, d, 32);
      float M = runM[v];
      float nM = fmaxf(M, m);
      runS[v] = runS[v] * __expf(M - nM) + s * __expf(m - nM);
      runM[v] = nM;
    }
  }

  if (ln == 0) {
#pragma unroll
    for (int v = 0; v < 8; ++v) {
      int row = rs * 16 + v + half_ * 8;
      float* p = mbuf + (cs * ROWS_PER_BLK + row) * 2;
      p[0] = runM[v];
      p[1] = runS[v];
    }
  }
  __syncthreads();

  if (threadIdx.x < ROWS_PER_BLK) {
    int row = threadIdx.x;
    float M0 = mbuf[row * 2],                  S0 = mbuf[row * 2 + 1];
    float M1 = mbuf[(ROWS_PER_BLK + row) * 2], S1 = mbuf[(ROWS_PER_BLK + row) * 2 + 1];
    float M = fmaxf(M0, M1);
    float S = S0 * __expf(M0 - M) + S1 * __expf(M1 - M);
    segM[seg * NROWS + rowbase + row] = M;
    segS[seg * NROWS + rowbase + row] = S;
  }
}

// -------- Fallback K2 (on-the-fly conversion) if workspace is too small ----
__global__ __launch_bounds__(256) void lse_gemm_fallback_kernel(
    const float* __restrict__ input, const float* __restrict__ weight,
    const float* __restrict__ bias,
    float* __restrict__ segM, float* __restrict__ segS)
{
  extern __shared__ uint32_t smem[];
  float* mbuf = (float*)(smem + ROWS_PER_BLK * AROW * 2);

  const int rowbase  = blockIdx.x * ROWS_PER_BLK;
  const int seg      = blockIdx.y;
  const int segStart = (seg == 0) ? 0    : (seg == 1 ? CUT0 : CUT1);
  const int segEnd   = (seg == 0) ? CUT0 : (seg == 1 ? CUT1 : NCLASS);
  const int nTiles   = (segEnd - segStart + 15) >> 4;

  uint32_t* AHI = smem;
  uint32_t* ALO = smem + ROWS_PER_BLK * AROW;
  for (int i = threadIdx.x; i < ROWS_PER_BLK * (IN_F / 2); i += blockDim.x) {
    int r  = i >> 9;
    int d  = i & 511;
    int blk = d >> 4, dl = d & 15;
    int jl = dl;
    const float* src = input + (size_t)(rowbase + r) * IN_F + blk * 32 + dl * 2;
    float x0 = src[0], x1 = src[1];
    _Float16 h0 = (_Float16)x0, h1 = (_Float16)x1;
    _Float16 l0 = (_Float16)(x0 - (float)h0), l1 = (_Float16)(x1 - (float)h1);
    AHI[r * AROW + blk * 16 + jl] = pk2(h0, h1);
    ALO[r * AROW + blk * 16 + jl] = pk2(l0, l1);
  }
  __syncthreads();

  const int wave  = threadIdx.x >> 5;
  const int lane  = threadIdx.x & 31;
  const int rs    = wave >> 1;
  const int cs    = wave & 1;
  const int half_ = lane >> 4;
  const int ln    = lane & 15;
  const uint32_t arow = (uint32_t)(rs * 16 + ln) * AROW + half_ * 8;

  float runM[8], runS[8];
#pragma unroll
  for (int v = 0; v < 8; ++v) { runM[v] = -1e30f; runS[v] = 0.f; }

  for (int t = cs; t < nTiles; t += 2) {
    const int cbase = segStart + (t << 4);
    const int c  = cbase + ln;
    const int cc = (c < NCLASS) ? c : (NCLASS - 1);
    const float* wrow = weight + (size_t)cc * IN_F + half_ * 16;

    v8f accH = {0.f, 0.f, 0.f, 0.f, 0.f, 0.f, 0.f, 0.f};
    v8f accX = {0.f, 0.f, 0.f, 0.f, 0.f, 0.f, 0.f, 0.f};
    v8f accY = {0.f, 0.f, 0.f, 0.f, 0.f, 0.f, 0.f, 0.f};
#pragma unroll 2
    for (int kb = 0; kb < IN_F; kb += 32) {
      const v8h* ah = (const v8h*)(AHI + arow + (kb >> 1));
      const v8h* al = (const v8h*)(ALO + arow + (kb >> 1));
      v16h ahi = CAT16(ah[0], ah[1]);
      v16h alo = CAT16(al[0], al[1]);
      v16h bhi, blo;
      const float4* wp = (const float4*)(wrow + kb);
#pragma unroll
      for (int q = 0; q < 4; ++q) {
        float4 w4 = wp[q];
        float wsv[4] = {w4.x, w4.y, w4.z, w4.w};
#pragma unroll
        for (int j = 0; j < 4; ++j) {
          float x = wsv[j];
          _Float16 h = (_Float16)x;
          _Float16 l = (_Float16)(x - (float)h);
          bhi[q * 4 + j] = h;
          blo[q * 4 + j] = l;
        }
      }
      accX = __builtin_amdgcn_wmma_f32_16x16x32_f16(false, alo, false, bhi, (short)0, accX, false, false);
      accY = __builtin_amdgcn_wmma_f32_16x16x32_f16(false, ahi, false, blo, (short)0, accY, false, false);
      accH = __builtin_amdgcn_wmma_f32_16x16x32_f16(false, ahi, false, bhi, (short)0, accH, false, false);
    }

    const bool  valid = (c < segEnd);
    const float b = valid ? bias[c] : 0.f;
#pragma unroll
    for (int v = 0; v < 8; ++v) {
      float x = valid ? (accH[v] + (accX[v] + accY[v]) + b) : -INFINITY;
      float m = x;
#pragma unroll
      for (int d = 1; d < 16; d <<= 1) m = fmaxf(m, __shfl_xor(m, d, 32));
      float e = __expf(x - m);
      float s = e;
#pragma unroll
      for (int d = 1; d < 16; d <<= 1) s += __shfl_xor(s, d, 32);
      float M = runM[v];
      float nM = fmaxf(M, m);
      runS[v] = runS[v] * __expf(M - nM) + s * __expf(m - nM);
      runM[v] = nM;
    }
  }

  if (ln == 0) {
#pragma unroll
    for (int v = 0; v < 8; ++v) {
      int row = rs * 16 + v + half_ * 8;
      float* p = mbuf + (cs * ROWS_PER_BLK + row) * 2;
      p[0] = runM[v];
      p[1] = runS[v];
    }
  }
  __syncthreads();

  if (threadIdx.x < ROWS_PER_BLK) {
    int row = threadIdx.x;
    float M0 = mbuf[row * 2],                  S0 = mbuf[row * 2 + 1];
    float M1 = mbuf[(ROWS_PER_BLK + row) * 2], S1 = mbuf[(ROWS_PER_BLK + row) * 2 + 1];
    float M = fmaxf(M0, M1);
    float S = S0 * __expf(M0 - M) + S1 * __expf(M1 - M);
    segM[seg * NROWS + rowbase + row] = M;
    segS[seg * NROWS + rowbase + row] = S;
  }
}

// ---------------- K3: combine head + tail terms per row --------------------
__global__ __launch_bounds__(256) void combine_kernel(
    const int* __restrict__ target,
    const float* __restrict__ tl, const float* __restrict__ tv0, const float* __restrict__ tv1,
    const float* __restrict__ segM, const float* __restrict__ segS,
    float* __restrict__ out)
{
  int r = blockIdx.x * blockDim.x + threadIdx.x;
  if (r >= NROWS) return;
  float M0 = segM[r], S0 = segS[r];
  float a = tv0[r], b = tv1[r];
  float hM = fmaxf(M0, fmaxf(a, b));
  float hS = S0 * __expf(M0 - hM) + __expf(a - hM) + __expf(b - hM);
  float head_lse = hM + logf(hS);
  int t = target[r];
  float res;
  if (t < CUT0) {
    res = tl[r] - head_lse;
  } else {
    int c = (t < CUT1) ? 1 : 2;
    float lc  = segM[c * NROWS + r] + logf(segS[c * NROWS + r]);
    float tvc = (c == 1) ? a : b;
    res = (tvc - head_lse) + (tl[r] - lc);
  }
  out[r] = res;
}

// ---------------- K4: loss = mean(-output) ---------------------------------
__global__ __launch_bounds__(256) void loss_kernel(
    const float* __restrict__ out, float* __restrict__ loss)
{
  __shared__ float red[8];
  float s = 0.f;
  for (int r = threadIdx.x; r < NROWS; r += 256) s += out[r];
#pragma unroll
  for (int d = 1; d < 32; d <<= 1) s += __shfl_xor(s, d, 32);
  const int wave = threadIdx.x >> 5, lane = threadIdx.x & 31;
  if (lane == 0) red[wave] = s;
  __syncthreads();
  if (threadIdx.x == 0) {
    float a = 0.f;
#pragma unroll
    for (int i = 0; i < 8; ++i) a += red[i];
    loss[0] = -a / (float)NROWS;
  }
}

extern "C" void kernel_launch(void* const* d_in, const int* in_sizes, int n_in,
                              void* d_out, int out_size, void* d_ws, size_t ws_size,
                              hipStream_t stream) {
  const float* input  = (const float*)d_in[0];
  const int*   target = (const int*)d_in[1];
  const float* weight = (const float*)d_in[2];
  const float* bias   = (const float*)d_in[3];
  const float* tailv  = (const float*)d_in[4];
  const float* tailb  = (const float*)d_in[5];
  float* out = (float*)d_out;     // [NROWS] log-probs + [1] loss

  float* ws   = (float*)d_ws;
  float* tl   = ws;               // NROWS
  float* tv0  = ws + NROWS;       // NROWS
  float* tv1  = ws + 2 * NROWS;   // NROWS
  float* segM = ws + 3 * NROWS;   // 3*NROWS
  float* segS = ws + 6 * NROWS;   // 3*NROWS

  const size_t base_floats = (size_t)9 * NROWS;
  const size_t welems = (size_t)NCLASS * IN_F;
  const size_t need = base_floats * sizeof(float) + 2 * welems * sizeof(uint16_t);

  row_dots_kernel<<<NROWS, 256, 0, stream>>>(input, target, weight, bias,
                                             tailv, tailb, tl, tv0, tv1);

  const size_t smem = (size_t)(2 * ROWS_PER_BLK * AROW + 2 * ROWS_PER_BLK * 2) * sizeof(uint32_t);
  dim3 grid(NROWS / ROWS_PER_BLK, 3);

  if (ws_size >= need) {
    uint32_t* whi32 = (uint32_t*)(ws + base_floats);
    uint32_t* wlo32 = whi32 + welems / 2;
    const int n4 = (int)(welems / 4);
    split_weight_kernel<<<(n4 + 255) / 256, 256, 0, stream>>>(weight, whi32, wlo32, n4);
    lse_gemm_split_kernel<<<grid, 256, smem, stream>>>(
        input, (const _Float16*)whi32, (const _Float16*)wlo32, bias, segM, segS);
  } else {
    lse_gemm_fallback_kernel<<<grid, 256, smem, stream>>>(input, weight, bias, segM, segS);
  }

  combine_kernel<<<NROWS / 256, 256, 0, stream>>>(target, tl, tv0, tv1, segM, segS, out);
  loss_kernel<<<1, 256, 0, stream>>>(out, out + NROWS);
}
